// MultimodalMambaWithKANAttention_58643483459597
// MI455X (gfx1250) — matrix operand order repaired and checked by reading, not compiled, not run on hardware
//
#include <hip/hip_runtime.h>

typedef __attribute__((ext_vector_type(16))) __bf16 v16bf;
typedef __attribute__((ext_vector_type(8)))  __bf16 v8bf;
typedef __attribute__((ext_vector_type(8)))  float  v8f;
typedef unsigned short u16;
typedef long long i64;

// ---------- bf16 helpers (storage as u16, round-to-nearest-even) ----------
__device__ __forceinline__ u16 f2bf(float f) {
    union { float f; unsigned u; } v; v.f = f;
    unsigned u = v.u;
    u += 0x7fffu + ((u >> 16) & 1u);
    return (u16)(u >> 16);
}

__device__ __forceinline__ v16bf cat16(v8bf a, v8bf b) {
    return __builtin_shufflevector(a, b, 0,1,2,3,4,5,6,7,8,9,10,11,12,13,14,15);
}

// =====================================================================
// Generic batched bf16 WMMA GEMM:  C[M x N] (f32) = A[M x K] * B[K x N]
// A row-major (row stride As_r elements, K contiguous).
// B: K contiguous (Bs_r == 1 implied), column stride Bs_c.
// Batch z offsets: off = (z/zdiv)*Hi + (z%zdiv)*Lo (element units).
// Grid: (N/64, M/128, Z), block 256 (8 waves, each one 16x64 tile).
// Requirements: N%64==0, M%128==0, K%32==0; bases/strides 16B-aligned.
// Software-pipelined: fragments for k+32 are loaded before the wmmas of k,
// so each wmma waits on loads issued a full iteration earlier.
// =====================================================================
__global__ __launch_bounds__(256) void gemm_bf16_wmma(
    const u16* __restrict__ A, const u16* __restrict__ B, float* __restrict__ C,
    int Kd,
    i64 As_r, i64 Bs_c, i64 Cs_r,
    int zdiv, i64 aHi, i64 aLo, i64 bHi, i64 bLo, i64 cHi, i64 cLo)
{
    int wave = threadIdx.x >> 5;
    int lane = threadIdx.x & 31;
    int row0 = (blockIdx.y * 8 + wave) * 16;
    int col0 = blockIdx.x * 64;
    int z = blockIdx.z;
    i64 aoff = (i64)(z / zdiv) * aHi + (i64)(z % zdiv) * aLo;
    i64 boff = (i64)(z / zdiv) * bHi + (i64)(z % zdiv) * bLo;
    i64 coff = (i64)(z / zdiv) * cHi + (i64)(z % zdiv) * cLo;
    int rlocal = lane & 15;   // row-in-tile for A, col-in-tile for B/C
    int ksel   = lane >> 4;   // K-half select

    // A: lane reads rows of A; 16 consecutive K: k0 + ksel*16 + j
    const u16* Ap = A + aoff + (i64)(row0 + rlocal) * As_r + ksel * 16;
    // B: lane reads 16 consecutive K at fixed column: k0 + ksel*16 + j
    const u16* Bp0 = B + boff + (i64)ksel * 16 + (i64)(col0 + rlocal) * Bs_c;
    const u16* Bp1 = Bp0 + 16 * Bs_c;
    const u16* Bp2 = Bp0 + 32 * Bs_c;
    const u16* Bp3 = Bp0 + 48 * Bs_c;

    v8f zero = {};
    v8f acc0 = zero, acc1 = zero, acc2 = zero, acc3 = zero;

    // prologue: fragments for k0 = 0
    v16bf af = cat16(*(const v8bf*)(Ap),      *(const v8bf*)(Ap + 8));
    v16bf b0 = cat16(*(const v8bf*)(Bp0),     *(const v8bf*)(Bp0 + 8));
    v16bf b1 = cat16(*(const v8bf*)(Bp1),     *(const v8bf*)(Bp1 + 8));
    v16bf b2 = cat16(*(const v8bf*)(Bp2),     *(const v8bf*)(Bp2 + 8));
    v16bf b3 = cat16(*(const v8bf*)(Bp3),     *(const v8bf*)(Bp3 + 8));

    #pragma unroll 2
    for (int k0 = 0; k0 + 32 < Kd; k0 += 32) {
        int kn = k0 + 32;
        // prefetch next k-step's fragments before consuming current ones
        v16bf afn = cat16(*(const v8bf*)(Ap + kn),      *(const v8bf*)(Ap + kn + 8));
        v16bf b0n = cat16(*(const v8bf*)(Bp0 + kn),     *(const v8bf*)(Bp0 + kn + 8));
        v16bf b1n = cat16(*(const v8bf*)(Bp1 + kn),     *(const v8bf*)(Bp1 + kn + 8));
        v16bf b2n = cat16(*(const v8bf*)(Bp2 + kn),     *(const v8bf*)(Bp2 + kn + 8));
        v16bf b3n = cat16(*(const v8bf*)(Bp3 + kn),     *(const v8bf*)(Bp3 + kn + 8));
        acc0 = __builtin_amdgcn_wmma_f32_16x16x32_bf16(false, af, false, b0, (short)0, acc0, false, false);
        acc1 = __builtin_amdgcn_wmma_f32_16x16x32_bf16(false, af, false, b1, (short)0, acc1, false, false);
        acc2 = __builtin_amdgcn_wmma_f32_16x16x32_bf16(false, af, false, b2, (short)0, acc2, false, false);
        acc3 = __builtin_amdgcn_wmma_f32_16x16x32_bf16(false, af, false, b3, (short)0, acc3, false, false);
        af = afn; b0 = b0n; b1 = b1n; b2 = b2n; b3 = b3n;
    }
    // epilogue: last k-step
    acc0 = __builtin_amdgcn_wmma_f32_16x16x32_bf16(false, af, false, b0, (short)0, acc0, false, false);
    acc1 = __builtin_amdgcn_wmma_f32_16x16x32_bf16(false, af, false, b1, (short)0, acc1, false, false);
    acc2 = __builtin_amdgcn_wmma_f32_16x16x32_bf16(false, af, false, b2, (short)0, acc2, false, false);
    acc3 = __builtin_amdgcn_wmma_f32_16x16x32_bf16(false, af, false, b3, (short)0, acc3, false, false);

    float* Cp = C + coff + (i64)(row0 + ksel * 8) * Cs_r + col0 + rlocal;
    #pragma unroll
    for (int r = 0; r < 8; ++r) {            // D layout: M = r + 8*(lane/16), N = lane&15
        float* Cr = Cp + (i64)r * Cs_r;
        Cr[0]  = acc0[r];
        Cr[16] = acc1[r];
        Cr[32] = acc2[r];
        Cr[48] = acc3[r];
    }
}

// ---------------- seq = concat(proj(text), proj(img), proj(first), proj(last)) + PE
__global__ __launch_bounds__(256) void build_seq(
    const float* __restrict__ text, const float* __restrict__ img,
    const float* __restrict__ firsth, const float* __restrict__ lasth,
    const float* __restrict__ Wt, const float* __restrict__ bt,
    const float* __restrict__ Wi, const float* __restrict__ bi,
    const float* __restrict__ Wf, const float* __restrict__ bfc,
    const float* __restrict__ Wl, const float* __restrict__ bl,
    float* __restrict__ seq)
{
    int l = blockIdx.x, b = blockIdx.y, o = threadIdx.x;
    float acc;
    if (l == 0) {
        acc = bt[o];
        const float* x = text + (i64)b * 768; const float* w = Wt + (i64)o * 768;
        for (int c = 0; c < 768; ++c) acc += x[c] * w[c];
    } else if (l <= 509) {
        int n = l - 1;
        acc = bi[o];
        const float* x = img + (i64)b * 640 * 509 + n; const float* w = Wi + (i64)o * 640;
        for (int c = 0; c < 640; ++c) acc += x[(i64)c * 509] * w[c];
    } else if (l == 510) {
        acc = bfc[o];
        const float* x = firsth + (i64)b * 3584; const float* w = Wf + (i64)o * 3584;
        for (int c = 0; c < 3584; ++c) acc += x[c] * w[c];
    } else {
        acc = bl[o];
        const float* x = lasth + (i64)b * 3584; const float* w = Wl + (i64)o * 3584;
        for (int c = 0; c < 3584; ++c) acc += x[c] * w[c];
    }
    float e = __expf(-logf(10000.0f) * (float)(2 * (o >> 1)) / 256.0f);
    float ang = (float)l * e;
    acc += (o & 1) ? __cosf(ang) : __sinf(ang);
    seq[((i64)b * 512 + l) * 256 + o] = acc;
}

// ---------------- KAN combined activation matrix: A[m][0:256]=silu(x), A[m][256+8i+g]=basis
__global__ __launch_bounds__(256) void kan_basis(const float* __restrict__ seq, u16* __restrict__ A)
{
    int m = blockIdx.x, i = threadIdx.x;
    float x = seq[(i64)m * 256 + i];
    u16* row = A + (i64)m * 2304;
    row[i] = f2bf(x / (1.f + __expf(-x)));        // silu
    float gr[12];
    #pragma unroll
    for (int j = 0; j < 12; ++j) gr[j] = 0.4f * (float)(j - 3) - 1.f;
    float bb[11];
    #pragma unroll
    for (int j = 0; j < 11; ++j) bb[j] = (x >= gr[j] && x < gr[j + 1]) ? 1.f : 0.f;
    #pragma unroll
    for (int p = 1; p <= 3; ++p) {
        float inv = 1.f / (0.4f * (float)p);       // uniform grid: denom = 0.4*p
        for (int j = 0; j + p < 11; ++j)
            bb[j] = (x - gr[j]) * inv * bb[j] + (gr[j + p + 1] - x) * inv * bb[j + 1];
    }
    #pragma unroll
    for (int g = 0; g < 8; ++g) row[256 + i * 8 + g] = f2bf(bb[g]);
}

// ---------------- pack [base_w | spline_w] N-major: wc[o*2304 + r] (K contiguous for B)
__global__ __launch_bounds__(256) void pack_wc(const float* __restrict__ bw,
                                               const float* __restrict__ sw,
                                               u16* __restrict__ wc)
{
    int r = blockIdx.x, o = threadIdx.x;
    float v;
    if (r < 256) v = bw[(i64)o * 256 + r];
    else { int q = r - 256; int i = q >> 3, g = q & 7; v = sw[((i64)o * 256 + i) * 8 + g]; }
    wc[(i64)o * 2304 + r] = f2bf(v);
}

__global__ void cvt_f32_bf16(const float* __restrict__ x, u16* __restrict__ y, int n)
{
    int i = blockIdx.x * 256 + threadIdx.x;
    if (i < n) y[i] = f2bf(x[i]);
}

// like cvt, but zero-fills [n_valid, n_total) (for N-padded weight buffers)
__global__ void cvt_f32_bf16_pad(const float* __restrict__ x, u16* __restrict__ y,
                                 int n_valid, int n_total)
{
    int i = blockIdx.x * 256 + threadIdx.x;
    if (i < n_total) y[i] = (i < n_valid) ? f2bf(x[i]) : (u16)0;
}

// ---------------- V transpose to K-contiguous layout: vt[(z*64+d)*512 + m]
__global__ __launch_bounds__(512) void transpose_v(const float* __restrict__ vf, u16* __restrict__ vt)
{
    int row = blockIdx.x;            // z*64 + d,  z = b*4+h
    int m = threadIdx.x;
    int z = row >> 6, d = row & 63;
    int b = z >> 2, h = z & 3;
    vt[(i64)row * 512 + m] = f2bf(vf[((i64)b * 512 + m) * 256 + h * 64 + d]);
}

// ---------------- softmax over 512 cols, scale 1/sqrt(64); bf16 output
__global__ __launch_bounds__(256) void softmax512(const float* __restrict__ S, u16* __restrict__ P)
{
    __shared__ float red[256];
    int row = blockIdx.x, t = threadIdx.x;
    const float sc = 0.125f;
    float v0 = S[(i64)row * 512 + t] * sc;
    float v1 = S[(i64)row * 512 + 256 + t] * sc;
    red[t] = fmaxf(v0, v1); __syncthreads();
    for (int s = 128; s > 0; s >>= 1) { if (t < s) red[t] = fmaxf(red[t], red[t + s]); __syncthreads(); }
    float m = red[0]; __syncthreads();
    float e0 = __expf(v0 - m), e1 = __expf(v1 - m);
    red[t] = e0 + e1; __syncthreads();
    for (int s = 128; s > 0; s >>= 1) { if (t < s) red[t] += red[t + s]; __syncthreads(); }
    float inv = 1.f / red[0];
    P[(i64)row * 512 + t]       = f2bf(e0 * inv);
    P[(i64)row * 512 + 256 + t] = f2bf(e1 * inv);
}

// ---------------- layernorm over 256 cols: LN(X + R + bias) * g + be
__global__ __launch_bounds__(256) void ln256(const float* __restrict__ X, const float* __restrict__ R,
                                             const float* __restrict__ bias,
                                             const float* __restrict__ g, const float* __restrict__ be,
                                             float* __restrict__ out, u16* __restrict__ obf)
{
    __shared__ float red[256];
    int row = blockIdx.x, t = threadIdx.x;
    float v = X[(i64)row * 256 + t];
    if (R)    v += R[(i64)row * 256 + t];
    if (bias) v += bias[t];
    red[t] = v; __syncthreads();
    for (int s = 128; s > 0; s >>= 1) { if (t < s) red[t] += red[t + s]; __syncthreads(); }
    float mean = red[0] * (1.f / 256.f); __syncthreads();
    float d = v - mean;
    red[t] = d * d; __syncthreads();
    for (int s = 128; s > 0; s >>= 1) { if (t < s) red[t] += red[t + s]; __syncthreads(); }
    float var = red[0] * (1.f / 256.f);
    float y = d * rsqrtf(var + 1e-5f) * g[t] + be[t];
    out[(i64)row * 256 + t] = y;
    if (obf) obf[(i64)row * 256 + t] = f2bf(y);
}

// ---------------- causal depthwise conv (K=4) + silu on xz[:, :512]
__global__ __launch_bounds__(512) void conv_silu(const float* __restrict__ xz,
                                                 const float* __restrict__ cw, const float* __restrict__ cb,
                                                 float* __restrict__ uf, u16* __restrict__ ub)
{
    int m = blockIdx.x, d = threadIdx.x;
    int l = m & 511;
    float acc = cb[d];
    #pragma unroll
    for (int j = 0; j < 4; ++j) {
        int ls = l - 3 + j;
        if (ls >= 0) acc += xz[((i64)m - 3 + j) * 1024 + d] * cw[d * 4 + j];
    }
    float u = acc / (1.f + __expf(-acc));
    uf[(i64)m * 512 + d] = u;
    ub[(i64)m * 512 + d] = f2bf(u);
}

// ---------------- dt = softplus(dbc[:, :16] @ dt_w.T + dt_b)   (dbc row stride 320)
__global__ __launch_bounds__(512) void dt_kernel(const float* __restrict__ dbc,
                                                 const float* __restrict__ dtw, const float* __restrict__ dtb,
                                                 float* __restrict__ dt)
{
    int m = blockIdx.x, d = threadIdx.x;
    float acc = dtb[d];
    const float* r = dbc + (i64)m * 320;
    const float* w = dtw + (i64)d * 16;
    #pragma unroll
    for (int j = 0; j < 16; ++j) acc += r[j] * w[j];
    dt[(i64)m * 512 + d] = (acc > 20.f) ? acc : log1pf(__expf(acc));
}

// ---------------- selective scan: one wave per (b,d), 4 states/lane, shfl reduce
__global__ __launch_bounds__(128) void mamba_scan(const float* __restrict__ dt, const float* __restrict__ u,
                                                  const float* __restrict__ dbc,
                                                  const float* __restrict__ A_log, const float* __restrict__ Dp,
                                                  float* __restrict__ y)
{
    int wave = threadIdx.x >> 5, lane = threadIdx.x & 31;
    int d = blockIdx.x * 4 + wave;
    int b = blockIdx.y;
    float Av[4], h[4];
    #pragma unroll
    for (int i = 0; i < 4; ++i) {
        int s = lane + 32 * i;
        Av[i] = -__expf(A_log[(i64)d * 128 + s]);
        h[i] = 0.f;
    }
    float dpv = Dp[d];
    for (int l = 0; l < 512; ++l) {
        i64 m = (i64)b * 512 + l;
        float dtv = dt[m * 512 + d];
        float uv  = u[m * 512 + d];
        float du  = dtv * uv;
        const float* bc = dbc + m * 320;          // padded row stride
        float acc = 0.f;
        #pragma unroll
        for (int i = 0; i < 4; ++i) {
            int s = lane + 32 * i;
            h[i] = h[i] * __expf(dtv * Av[i]) + du * bc[16 + s];   // B at cols 16..143
            acc += h[i] * bc[144 + s];                             // C at cols 144..271
        }
        #pragma unroll
        for (int off = 16; off > 0; off >>= 1) acc += __shfl_xor(acc, off);
        if (lane == 0) y[m * 512 + d] = acc + dpv * uv;
    }
}

// ---------------- gate: g = bf16( y * silu(z) ),  z = xz[:, 512:]
__global__ __launch_bounds__(512) void gate_kernel(const float* __restrict__ xz, const float* __restrict__ y,
                                                   u16* __restrict__ g)
{
    int m = blockIdx.x, d = threadIdx.x;
    float z = xz[(i64)m * 1024 + 512 + d];
    float sz = z / (1.f + __expf(-z));
    g[(i64)m * 512 + d] = f2bf(y[(i64)m * 512 + d] * sz);
}

// =====================================================================
extern "C" void kernel_launch(void* const* d_in, const int* in_sizes, int n_in,
                              void* d_out, int out_size, void* d_ws, size_t ws_size,
                              hipStream_t stream)
{
    (void)in_sizes; (void)n_in; (void)out_size; (void)ws_size;
    const float* text   = (const float*)d_in[0];
    const float* img    = (const float*)d_in[1];
    const float* firsth = (const float*)d_in[2];
    const float* lasth  = (const float*)d_in[3];
    const float* W_text = (const float*)d_in[4];  const float* b_text = (const float*)d_in[5];
    const float* W_img  = (const float*)d_in[6];  const float* b_img  = (const float*)d_in[7];
    const float* W_first= (const float*)d_in[8];  const float* b_first= (const float*)d_in[9];
    const float* W_last = (const float*)d_in[10]; const float* b_last = (const float*)d_in[11];
    const float* q_base = (const float*)d_in[12]; const float* q_spline = (const float*)d_in[13];
    const float* k_base = (const float*)d_in[14]; const float* k_spline = (const float*)d_in[15];
    const float* v_base = (const float*)d_in[16]; const float* v_spline = (const float*)d_in[17];
    const float* W_ao   = (const float*)d_in[18]; const float* b_ao   = (const float*)d_in[19];
    const float* g1     = (const float*)d_in[20]; const float* be1    = (const float*)d_in[21];
    const float* g2     = (const float*)d_in[22]; const float* be2    = (const float*)d_in[23];
    const float* in_proj= (const float*)d_in[24];
    const float* conv_w = (const float*)d_in[25]; const float* conv_b = (const float*)d_in[26];
    const float* x_proj = (const float*)d_in[27];
    const float* dt_w   = (const float*)d_in[28]; const float* dt_b   = (const float*)d_in[29];
    const float* A_log  = (const float*)d_in[30]; const float* D_p    = (const float*)d_in[31];
    const float* out_proj=(const float*)d_in[32];

    // --- workspace bump allocator (256B aligned) ---
    char* wsp = (char*)d_ws;
    size_t off = 0;
    auto alloc = [&](size_t bytes) -> void* {
        void* p = wsp + off;
        off = (off + bytes + 255) & ~(size_t)255;
        return p;
    };
    const size_t M4 = 4096;   // B*L rows
    float* seq  = (float*)alloc(M4 * 256 * 4);
    u16*   Akan = (u16*)  alloc(M4 * 2304 * 2);
    u16*   wcq  = (u16*)  alloc(2304 * 256 * 2);
    u16*   wck  = (u16*)  alloc(2304 * 256 * 2);
    u16*   wcv  = (u16*)  alloc(2304 * 256 * 2);
    float* qf   = (float*)alloc(M4 * 256 * 4);
    float* kf   = (float*)alloc(M4 * 256 * 4);
    float* vf   = (float*)alloc(M4 * 256 * 4);
    u16*   qb   = (u16*)  alloc(M4 * 256 * 2);
    u16*   kb   = (u16*)  alloc(M4 * 256 * 2);
    u16*   vt   = (u16*)  alloc((size_t)32 * 64 * 512 * 2);
    float* scor = (float*)alloc((size_t)32 * 512 * 512 * 4);
    u16*   Pm   = (u16*)  alloc((size_t)32 * 512 * 512 * 2);
    float* Of   = (float*)alloc(M4 * 256 * 4);
    u16*   Ob   = (u16*)  alloc(M4 * 256 * 2);
    u16*   waob = (u16*)  alloc(256 * 256 * 2);
    float* attn = (float*)alloc(M4 * 256 * 4);
    float* af_  = (float*)alloc(M4 * 256 * 4);
    u16*   ab_  = (u16*)  alloc(M4 * 256 * 2);
    u16*   ipb  = (u16*)  alloc(1024 * 256 * 2);
    float* xz   = (float*)alloc(M4 * 1024 * 4);
    float* uf   = (float*)alloc(M4 * 512 * 4);
    u16*   ub   = (u16*)  alloc(M4 * 512 * 2);
    u16*   xpb  = (u16*)  alloc(320 * 512 * 2);      // N padded 272 -> 320
    float* dbc  = (float*)alloc(M4 * 320 * 4);       // row stride 320
    float* dtv  = (float*)alloc(M4 * 512 * 4);
    float* yv   = (float*)alloc(M4 * 512 * 4);
    u16*   gb   = (u16*)  alloc(M4 * 512 * 2);
    u16*   opb  = (u16*)  alloc(512 * 256 * 2);
    float* mf   = (float*)alloc(M4 * 256 * 4);

    auto gemm = [&](const void* A, const void* B, void* C, int M, int N, int K,
                    i64 As_r, i64 Bs_c, i64 Cs_r,
                    int Z, int zdiv, i64 aHi, i64 aLo, i64 bHi, i64 bLo, i64 cHi, i64 cLo) {
        dim3 grid(N / 64, M / 128, Z);
        gemm_bf16_wmma<<<grid, 256, 0, stream>>>((const u16*)A, (const u16*)B, (float*)C,
            K, As_r, Bs_c, Cs_r, zdiv, aHi, aLo, bHi, bLo, cHi, cLo);
    };
    auto cvt = [&](const float* x, u16* y, int n) {
        cvt_f32_bf16<<<(n + 255) / 256, 256, 0, stream>>>(x, y, n);
    };

    // 1) embeddings + positional encoding
    build_seq<<<dim3(512, 8), 256, 0, stream>>>(text, img, firsth, lasth,
        W_text, b_text, W_img, b_img, W_first, b_first, W_last, b_last, seq);

    // 2) KAN: shared basis matrix + packed (N-major) weights; 3 WMMA GEMMs (K=2304)
    kan_basis<<<4096, 256, 0, stream>>>(seq, Akan);
    pack_wc<<<2304, 256, 0, stream>>>(q_base, q_spline, wcq);
    pack_wc<<<2304, 256, 0, stream>>>(k_base, k_spline, wck);
    pack_wc<<<2304, 256, 0, stream>>>(v_base, v_spline, wcv);
    gemm(Akan, wcq, qf, 4096, 256, 2304, 2304, 2304, 256, 1, 1, 0,0,0,0,0,0);
    gemm(Akan, wck, kf, 4096, 256, 2304, 2304, 2304, 256, 1, 1, 0,0,0,0,0,0);
    gemm(Akan, wcv, vf, 4096, 256, 2304, 2304, 2304, 256, 1, 1, 0,0,0,0,0,0);
    cvt(qf, qb, 4096 * 256); cvt(kf, kb, 4096 * 256);
    transpose_v<<<2048, 512, 0, stream>>>(vf, vt);

    // 3) attention: scores = Q K^T (batched over z = b*4+h), softmax, O = P V
    gemm(qb, kb, scor, 512, 512, 64, 256, 256, 512,
         32, 4, 131072, 64, 131072, 64, 1048576, 262144);
    softmax512<<<32 * 512, 256, 0, stream>>>(scor, Pm);
    gemm(Pm, vt, Of, 512, 64, 512, 512, 512, 256,
         32, 4, 1048576, 262144, 131072, 32768, 131072, 64);
    cvt(Of, Ob, 4096 * 256);
    cvt(W_ao, waob, 256 * 256);
    gemm(Ob, waob, attn, 4096, 256, 256, 256, 256, 256, 1, 1, 0,0,0,0,0,0);

    // 4) a = LN(attn + b_ao + seq)
    ln256<<<4096, 256, 0, stream>>>(attn, seq, b_ao, g1, be1, af_, ab_);

    // 5) Mamba
    cvt(in_proj, ipb, 1024 * 256);
    gemm(ab_, ipb, xz, 4096, 1024, 256, 256, 256, 1024, 1, 1, 0,0,0,0,0,0);
    conv_silu<<<4096, 512, 0, stream>>>(xz, conv_w, conv_b, uf, ub);
    // x_proj packed + zero-padded from N=272 to N=320 (branchless GEMM tiles)
    cvt_f32_bf16_pad<<<(320 * 512 + 255) / 256, 256, 0, stream>>>(x_proj, xpb, 272 * 512, 320 * 512);
    gemm(ub, xpb, dbc, 4096, 320, 512, 512, 512, 320, 1, 1, 0,0,0,0,0,0);
    dt_kernel<<<4096, 512, 0, stream>>>(dbc, dt_w, dt_b, dtv);
    mamba_scan<<<dim3(128, 8), 128, 0, stream>>>(dtv, uf, dbc, A_log, D_p, yv);
    gate_kernel<<<4096, 512, 0, stream>>>(xz, yv, gb);
    cvt(out_proj, opb, 512 * 256);
    gemm(gb, opb, mf, 4096, 256, 512, 512, 512, 256, 1, 1, 0,0,0,0,0,0);

    // 6) out = LN(m + a)
    ln256<<<4096, 256, 0, stream>>>(mf, af_, nullptr, g2, be2, (float*)d_out, nullptr);
}
